// RLagnet_54726473285944
// MI455X (gfx1250) — matrix operand order repaired and checked by reading, not compiled, not run on hardware
//
#include <hip/hip_runtime.h>
#include <hip/hip_bf16.h>

typedef __attribute__((ext_vector_type(16))) __bf16 v16bf;
typedef __attribute__((ext_vector_type(8)))  float  v8f;

#define IN_CHS 512
#define RED 128
#define ROWS_PER_BLK 128
#define KCHUNK 128

#if defined(__has_builtin) && __has_builtin(__builtin_amdgcn_sched_group_barrier)
#define SCHED_GROUP(mask, cnt, id) __builtin_amdgcn_sched_group_barrier(mask, cnt, id)
#else
#define SCHED_GROUP(mask, cnt, id)
#endif

union ABFrag { v16bf v; unsigned int u[8]; };

// hardware RNE f32->bf16 (backend emits packed cvt)
__device__ __forceinline__ unsigned int pk_bf16(float lo, float hi) {
  union { __bf16 h[2]; unsigned int u; } p;
  p.h[0] = (__bf16)lo;
  p.h[1] = (__bf16)hi;
  return p.u;
}
__device__ __forceinline__ unsigned short bf16_1(float f) {
  union { __bf16 h; unsigned short s; } p;
  p.h = (__bf16)f;
  return p.s;
}
__device__ __forceinline__ float hash_u01(unsigned int x) {
  x ^= x >> 17; x *= 0xed5ad4bbu;
  x ^= x >> 11; x *= 0xac4c1b51u;
  x ^= x >> 15; x *= 0x31848babu;
  x ^= x >> 14;
  return (float)(x >> 8) * (1.0f / 16777216.0f) + 1.0e-7f;
}

__global__ __launch_bounds__(256) void mlp_head_gumbel_kernel(
    const float* __restrict__ x, const float* __restrict__ W1,
    const float* __restrict__ b1, const float* __restrict__ W2,
    const float* __restrict__ b2, int* __restrict__ out) {
  // 64KB staging: x-tile bf16 [128][128] (32KB) + W1^T bf16 [128][128] (32KB),
  // later re-used as f32 h-buffer [128][128] (64KB).
  __shared__ __align__(16) unsigned char smem[65536];
  __shared__ float w2s[2 * RED + 2];

  unsigned int*   xs_u  = (unsigned int*)smem;             // [r=128][64] bf16x2
  unsigned short* wsT_s = (unsigned short*)(smem + 32768); // [n=128][k=128]
  const unsigned int* wsT_u = (const unsigned int*)(smem + 32768);
  float* hbuf = (float*)smem;                              // [r=128][c=128]

  const int t = threadIdx.x;
  const int lane = t & 31;
  const int wave = t >> 5;
  const int row0 = blockIdx.x * ROWS_PER_BLK;
  const int mbase = wave * 16;
  const int halfsel = lane >> 4;  // lane-half selects K sub-range
  const int l16 = lane & 15;

  if (t < 2 * RED) w2s[t] = W2[t];
  if (t < 2)       w2s[2 * RED + t] = b2[t];

  v8f acc[8] = {};

  const float4* x4 = (const float4*)x;
  const float4* w4 = (const float4*)W1;

  for (int kc = 0; kc < IN_CHS / KCHUNK; ++kc) {
    const int k0 = kc * KCHUNK;
    __syncthreads();  // previous chunk's consumers done before overwrite

    // ---- stage x tile: 128 rows x 128 k, f32 -> bf16 (16 float4 / thread)
    #pragma unroll
    for (int i = 0; i < 16; ++i) {
      int s  = t + i * 256;   // 0..4095 float4 slots
      int r  = s >> 5;        // 32 float4 per row
      int kq = s & 31;
      float4 v = x4[(size_t)(row0 + r) * (IN_CHS / 4) + (k0 >> 2) + kq];
      xs_u[r * 64 + kq * 2 + 0] = pk_bf16(v.x, v.y);
      xs_u[r * 64 + kq * 2 + 1] = pk_bf16(v.z, v.w);
    }
    // ---- stage W1 chunk transposed: wsT[n][k] = W1[k0+k][n], bf16
    #pragma unroll
    for (int i = 0; i < 16; ++i) {
      int s  = t + i * 256;
      int k  = s >> 5;
      int nq = s & 31;
      float4 v = w4[(size_t)(k0 + k) * (RED / 4) + nq];
      int n = nq * 4;
      wsT_s[(n + 0) * KCHUNK + k] = bf16_1(v.x);
      wsT_s[(n + 1) * KCHUNK + k] = bf16_1(v.y);
      wsT_s[(n + 2) * KCHUNK + k] = bf16_1(v.z);
      wsT_s[(n + 3) * KCHUNK + k] = bf16_1(v.w);
    }
    __syncthreads();

    // prefetch next x chunk into cache while WMMA runs
    if (kc + 1 < IN_CHS / KCHUNK) {
      const float* nxt =
          x + (size_t)(row0 + (t >> 1)) * IN_CHS + k0 + KCHUNK + (t & 1) * 64;
      __builtin_prefetch(nxt, 0, 0);
    }

    // ---- bf16 WMMA: 16 rows x 128 cols per wave, K step 32.
    // Per K-step: pin a scheduling group of 18 ds_reads (A: 2xb128,
    // B: 8 frags x 2xb128) followed by 8 back-to-back WMMAs so the XDL
    // pipe is fed without a ds wait between every matrix op.
    const int arow = mbase + l16;
    #pragma unroll
    for (int kk = 0; kk < 4; ++kk) {
      ABFrag a;
      #pragma unroll
      for (int j = 0; j < 4; ++j) {  // A 16x32: v0..3 K=0..7|8..15, v4..7 K=16..23|24..31
        a.u[j]     = xs_u[arow * 64 + kk * 16 + halfsel * 4 + j];
        a.u[4 + j] = xs_u[arow * 64 + kk * 16 + 8 + halfsel * 4 + j];
      }
      ABFrag bfr[8];
      #pragma unroll
      for (int nt = 0; nt < 8; ++nt) {
        const int n = nt * 16 + l16;
        #pragma unroll
        for (int j = 0; j < 8; ++j)  // B 32x16: lane=N, K pairs; v0..3 K=0..7|8..15, v4..7 K=16..23|24..31
          bfr[nt].u[j] = wsT_u[n * 64 + kk * 16 + halfsel * 4 + (j >> 2) * 4 + j];
      }
      #pragma unroll
      for (int nt = 0; nt < 8; ++nt)
        acc[nt] = __builtin_amdgcn_wmma_f32_16x16x32_bf16(
            false, a.v, false, bfr[nt].v, (short)0, acc[nt], false, false);
      SCHED_GROUP(0x0100, 18, 0);  // 18 DS reads first
      SCHED_GROUP(0x0008, 8, 0);   // then 8 WMMAs back-to-back
    }
  }

  __syncthreads();
  // ---- bias + ReLU, spill h to LDS (aliases staging buffers)
  #pragma unroll
  for (int nt = 0; nt < 8; ++nt) {
    const int col = nt * 16 + l16;
    const float bias = b1[col];
    #pragma unroll
    for (int i = 0; i < 8; ++i) {   // C 16x16: vgpr i -> M = i + 8*lanehalf
      int r = mbase + i + halfsel * 8;
      float hv = acc[nt][i] + bias;
      hbuf[r * RED + col] = hv > 0.f ? hv : 0.f;
    }
  }
  __syncthreads();

  // ---- layer 2 + Gumbel-max (log_softmax not needed: argmax invariant)
  if (t < ROWS_PER_BLK) {
    float l0 = w2s[2 * RED + 0], l1 = w2s[2 * RED + 1];
    #pragma unroll 4
    for (int i = 0; i < RED; ++i) {
      int c = (i + (t & 31)) & (RED - 1);  // stagger LDS banks
      float hv = hbuf[t * RED + c];
      l0 = fmaf(hv, w2s[2 * c + 0], l0);
      l1 = fmaf(hv, w2s[2 * c + 1], l1);
    }
    unsigned int grow = (unsigned int)(row0 + t);
    float g0 = -__logf(-__logf(hash_u01(grow * 2u + 0x9e3779b9u)));
    float g1 = -__logf(-__logf(hash_u01(grow * 2u + 1u + 0x7f4a7c15u)));
    out[grow] = (l1 + g1 > l0 + g0) ? 1 : 0;
  }
}

extern "C" void kernel_launch(void* const* d_in, const int* in_sizes, int n_in,
                              void* d_out, int out_size, void* d_ws, size_t ws_size,
                              hipStream_t stream) {
  const float* x  = (const float*)d_in[0];
  const float* W1 = (const float*)d_in[1];
  const float* b1 = (const float*)d_in[2];
  const float* W2 = (const float*)d_in[3];
  const float* b2 = (const float*)d_in[4];
  int batch  = in_sizes[0] / IN_CHS;          // 131072
  int blocks = batch / ROWS_PER_BLK;          // 1024
  mlp_head_gumbel_kernel<<<blocks, 256, 0, stream>>>(x, W1, b1, W2, b2,
                                                     (int*)d_out);
}
